// PhysicsInformedGRU_13168369729789
// MI455X (gfx1250) — compile-verified, matrix operand order checked by reading, not run on hardware
//
#include <hip/hip_runtime.h>
#include <hip/hip_bf16.h>

#define BATCH 1024
#define SEQ   512
#define HID   128

typedef __attribute__((ext_vector_type(16))) __bf16 v16bf;
typedef __attribute__((ext_vector_type(8)))  float  v8f;
typedef __attribute__((ext_vector_type(4)))  unsigned int v4u;

union V16 {
    v16bf v;
    unsigned short u[16];
    v4u q[2];
};

// Hardware f32 -> bf16 (v_cvt), instead of software RNE bit-twiddling.
__device__ __forceinline__ unsigned short f2bf(float f) {
    union { __bf16 b; unsigned short s; } u;
    u.b = (__bf16)f;
    return u.s;
}
__device__ __forceinline__ float bf2f(unsigned short s) {
    return __uint_as_float(((unsigned int)s) << 16);
}

// Load one 16x32 bf16 B-chunk (K x N, B[k][n] = W[n][k]) from row-major f32 W[*][128].
// Lane layout mirrors the A layout: lanes hold N=ln; VGPR v0-3: K=kbase+8*ksel+{0..7},
// VGPR v4-7: K=kbase+16+8*ksel+{0..7}.
__device__ __forceinline__ v16bf load_B_f32(const float* __restrict__ W, int row,
                                            int kbase, int ksel, bool valid) {
    V16 x;
#pragma unroll
    for (int i = 0; i < 8; ++i) {
        float a = valid ? W[row * HID + kbase + 8 * ksel + i] : 0.0f;
        float b = valid ? W[row * HID + kbase + 16 + 8 * ksel + i] : 0.0f;
        x.u[i]     = f2bf(a);
        x.u[8 + i] = f2bf(b);
    }
    return x.v;
}

// Load one 16x32 bf16 A-chunk of h from LDS (row stride 136 halves, 16B aligned rows).
__device__ __forceinline__ v16bf load_A(const unsigned short* __restrict__ h_lds,
                                        int m, int kbase, int ksel) {
    V16 x;
    x.q[0] = *(const v4u*)&h_lds[m * 136 + kbase + 8 * ksel];       // K +0..7 (half ksel)
    x.q[1] = *(const v4u*)&h_lds[m * 136 + kbase + 16 + 8 * ksel];  // K +16..23 (half ksel)
    return x.v;
}

#define WMMA_BF16(A, B, C) \
    __builtin_amdgcn_wmma_f32_16x16x32_bf16(false, (A), false, (B), (short)0, (C), false, false)

// ---------------------------------------------------------------------------
// Kernel 1: fused GRU scan + output head.
// Grid: BATCH/16 workgroups of 256 threads (8 waves). Wave w owns hidden slice
// j in [16w, 16w+16) and all three gates for it (12 WMMAs/step). Wave 0 also
// runs the x_pred head (4 WMMAs) on the previous step's A registers.
// ---------------------------------------------------------------------------
__global__ __launch_bounds__(256)
void gru_scan_kernel(const float* __restrict__ v_seq,   // [B,S,2]
                     const float* __restrict__ W_ih,    // [384,2]
                     const float* __restrict__ W_hh,    // [384,128]
                     const float* __restrict__ b_ih,    // [384]
                     const float* __restrict__ b_hh,    // [384]
                     const float* __restrict__ W_out,   // [2,128]
                     const float* __restrict__ b_out,   // [2]
                     float* __restrict__ x_pred)        // [B,S,2]
{
    __shared__ __align__(16) unsigned short h_lds[16 * 136];  // bf16 h[16][128], padded

    const int tid  = threadIdx.x;
    const int lane = tid & 31;
    const int wave = tid >> 5;       // 0..7
    const int ln   = lane & 15;      // N / M index within tile
    const int ksel = lane >> 4;      // half-wave select
    const int jb   = wave * 16;      // hidden slice base
    const int j    = jb + ln;        // this lane's hidden column
    const int bbase = blockIdx.x * 16;

    // zero-init h0
    for (int i = tid; i < 16 * 136; i += 256) h_lds[i] = 0;

    // per-lane scalar weights for input projection + hidden bias (gates r,z,n)
    float wih0[3], wih1[3], bihv[3], bhhv[3];
#pragma unroll
    for (int g = 0; g < 3; ++g) {
        int row = g * HID + j;
        wih0[g] = W_ih[row * 2 + 0];
        wih1[g] = W_ih[row * 2 + 1];
        bihv[g] = b_ih[row];
        bhhv[g] = b_hh[row];
    }

    // W_hh in register B-layout: 3 gates x 4 K-chunks (96 VGPRs, loaded once)
    v16bf Bh[3][4];
#pragma unroll
    for (int g = 0; g < 3; ++g)
#pragma unroll
        for (int kc = 0; kc < 4; ++kc)
            Bh[g][kc] = load_B_f32(W_hh, g * HID + j, 32 * kc, ksel, true);

    // W_out padded 2->16 columns, B-layout
    v16bf Bo[4];
#pragma unroll
    for (int kc = 0; kc < 4; ++kc)
        Bo[kc] = load_B_f32(W_out, ln, 32 * kc, ksel, ln < 2);
    const float bo = (ln < 2) ? b_out[ln] : 0.0f;

    __syncthreads();

#pragma unroll 1
    for (int t = 0; t < SEQ; ++t) {
        // ---- read phase: h_{t-1} from LDS ----
        v16bf A[4];
#pragma unroll
        for (int kc = 0; kc < 4; ++kc) A[kc] = load_A(h_lds, ln, 32 * kc, ksel);

        float hold[8];
#pragma unroll
        for (int r = 0; r < 8; ++r) {
            int m = 8 * ksel + r;
            hold[r] = bf2f(h_lds[m * 136 + j]);
        }

        // output head for step t-1 (wave 0; A regs hold h_{t-1})
        if (wave == 0 && t > 0) {
            v8f px = {0.f, 0.f, 0.f, 0.f, 0.f, 0.f, 0.f, 0.f};
#pragma unroll
            for (int kc = 0; kc < 4; ++kc) px = WMMA_BF16(A[kc], Bo[kc], px);
            if (ln < 2) {
#pragma unroll
                for (int r = 0; r < 8; ++r) {
                    int m = 8 * ksel + r;
                    size_t b = (size_t)(bbase + m);
                    x_pred[(b * SEQ + (t - 1)) * 2 + ln] = px[r] + bo;
                }
            }
        }
        __syncthreads();   // all reads of h_{t-1} complete

        // ---- recurrent matmul: gh for this wave's slice, 3 gates ----
        v8f D[3];
#pragma unroll
        for (int g = 0; g < 3; ++g) {
            v8f acc = {0.f, 0.f, 0.f, 0.f, 0.f, 0.f, 0.f, 0.f};
#pragma unroll
            for (int kc = 0; kc < 4; ++kc) acc = WMMA_BF16(A[kc], Bh[g][kc], acc);
            D[g] = acc;
        }

        // ---- gate nonlinearity + state update ----
#pragma unroll
        for (int r = 0; r < 8; ++r) {
            int m = 8 * ksel + r;
            const float* vp = v_seq + ((size_t)(bbase + m) * SEQ + t) * 2;
            float v0 = vp[0], v1 = vp[1];
            __builtin_prefetch(vp + 16, 0, 0);   // prefetch 8 steps ahead (global_prefetch_b8)

            float xr = wih0[0] * v0 + wih1[0] * v1 + bihv[0];
            float xz = wih0[1] * v0 + wih1[1] * v1 + bihv[1];
            float xn = wih0[2] * v0 + wih1[2] * v1 + bihv[2];
            float hr = D[0][r] + bhhv[0];
            float hz = D[1][r] + bhhv[1];
            float hn = D[2][r] + bhhv[2];

            float rg = 1.0f / (1.0f + __expf(-(xr + hr)));
            float zg = 1.0f / (1.0f + __expf(-(xz + hz)));
            float ng = tanhf(xn + rg * hn);
            float hnew = (1.0f - zg) * ng + zg * hold[r];

            h_lds[m * 136 + j] = f2bf(hnew);
        }
        __syncthreads();   // h_t published
    }

    // epilogue: x_pred for t = SEQ-1
    v16bf A2[4];
#pragma unroll
    for (int kc = 0; kc < 4; ++kc) A2[kc] = load_A(h_lds, ln, 32 * kc, ksel);
    if (wave == 0) {
        v8f px = {0.f, 0.f, 0.f, 0.f, 0.f, 0.f, 0.f, 0.f};
#pragma unroll
        for (int kc = 0; kc < 4; ++kc) px = WMMA_BF16(A2[kc], Bo[kc], px);
        if (ln < 2) {
#pragma unroll
            for (int r = 0; r < 8; ++r) {
                int m = 8 * ksel + r;
                size_t b = (size_t)(bbase + m);
                x_pred[(b * SEQ + (SEQ - 1)) * 2 + ln] = px[r] + bo;
            }
        }
    }
}

// ---------------------------------------------------------------------------
// Kernel 2: residual MLP + physics violations, fully parallel over (b, t).
// ---------------------------------------------------------------------------
__global__ __launch_bounds__(256)
void physics_kernel(const float* __restrict__ x0,      // [B,2]
                    const float* __restrict__ v_seq,   // [B,S,2]
                    const float* __restrict__ x_pred,  // [B,S,2]
                    const float* __restrict__ W_r1,    // [64,4]
                    const float* __restrict__ b_r1,    // [64]
                    const float* __restrict__ W_r2,    // [2,64]
                    const float* __restrict__ b_r2,    // [2]
                    float* __restrict__ viol)          // [B,S,2]
{
    __shared__ float sW1[64 * 4];
    __shared__ float sb1[64];
    __shared__ float sW2[2 * 64];
    __shared__ float sb2[2];

    int tid = threadIdx.x;
    if (tid < 256) sW1[tid] = W_r1[tid];
    if (tid < 64)  sb1[tid] = b_r1[tid];
    if (tid < 128) sW2[tid] = W_r2[tid];
    if (tid < 2)   sb2[tid] = b_r2[tid];
    __syncthreads();

    size_t idx = (size_t)blockIdx.x * blockDim.x + tid;   // over B*S
    int b = (int)(idx / SEQ);
    int t = (int)(idx % SEQ);

    float xp0, xp1;
    if (t == 0) {
        xp0 = x0[b * 2 + 0];
        xp1 = x0[b * 2 + 1];
    } else {
        size_t p = ((size_t)b * SEQ + (t - 1)) * 2;
        xp0 = x_pred[p];
        xp1 = x_pred[p + 1];
    }
    size_t c = ((size_t)b * SEQ + t) * 2;
    float v0 = v_seq[c], v1 = v_seq[c + 1];

    float r0 = sb2[0], r1 = sb2[1];
#pragma unroll 8
    for (int jj = 0; jj < 64; ++jj) {
        float h = sW1[jj * 4 + 0] * xp0 + sW1[jj * 4 + 1] * xp1 +
                  sW1[jj * 4 + 2] * v0 + sW1[jj * 4 + 3] * v1 + sb1[jj];
        h = fmaxf(h, 0.0f);
        r0 += sW2[jj] * h;
        r1 += sW2[64 + jj] * h;
    }
    viol[c]     = x_pred[c]     - (xp0 + v0 + r0);
    viol[c + 1] = x_pred[c + 1] - (xp1 + v1 + r1);
}

extern "C" void kernel_launch(void* const* d_in, const int* in_sizes, int n_in,
                              void* d_out, int out_size, void* d_ws, size_t ws_size,
                              hipStream_t stream) {
    const float* x0    = (const float*)d_in[0];
    const float* v_seq = (const float*)d_in[1];
    const float* W_ih  = (const float*)d_in[2];
    const float* W_hh  = (const float*)d_in[3];
    const float* b_ih  = (const float*)d_in[4];
    const float* b_hh  = (const float*)d_in[5];
    const float* W_out = (const float*)d_in[6];
    const float* b_out = (const float*)d_in[7];
    const float* W_r1  = (const float*)d_in[8];
    const float* b_r1  = (const float*)d_in[9];
    const float* W_r2  = (const float*)d_in[10];
    const float* b_r2  = (const float*)d_in[11];

    float* x_pred = (float*)d_out;                               // [B,S,2]
    float* viol   = (float*)d_out + (size_t)BATCH * SEQ * 2;     // [B,S,2]

    gru_scan_kernel<<<BATCH / 16, 256, 0, stream>>>(
        v_seq, W_ih, W_hh, b_ih, b_hh, W_out, b_out, x_pred);

    physics_kernel<<<(BATCH * SEQ) / 256, 256, 0, stream>>>(
        x0, v_seq, x_pred, W_r1, b_r1, W_r2, b_r2, viol);
}
